// RecNet_18459769438851
// MI455X (gfx1250) — matrix operand order repaired and checked by reading, not compiled, not run on hardware
//
#include <hip/hip_runtime.h>

// ---------------------------------------------------------------------------
// DWMTJ RecNet forward scan on MI455X (gfx1250), wave32 + WMMA bf16 +
// async global->LDS DMA (ASYNCcnt) for the serial-scan input stream.
//
//  Phase 0: swizzle w_in / w_rec / w_out (f32 -> bf16) into the per-lane
//           B-matrix fragment layout of V_WMMA_F32_16X16X32_BF16.
//  Phase 1: Ain = X @ w_in^T   ([25600 x 2312] x [2312 x 512]), bf16 WMMA,
//           f32 accumulate, result f32 in workspace. Memory-bound: 237 MB of
//           X @ 23.3 TB/s ~= 10 us floor.
//  Phase 2: 100-step scan, 16 self-contained workgroups (16 batch rows each).
//           z double-buffered in LDS -> single s_barrier per step. Each wave
//           streams its own 64-column Ain stripe for step t+1 with
//           GLOBAL_LOAD_ASYNC_TO_LDS_B128 while step t+1's 64 recurrent WMMAs
//           run; own-wave s_wait_asynccnt 0 is sufficient for visibility.
// ---------------------------------------------------------------------------

#define T_STEPS 100
#define BF      256
#define IN_F    2312
#define HID     512
#define OUTF    10

#define DT_C        1e-10f
#define W2_C        2.5e-8f
#define I_APP_C     1e-4f
#define MU_C        2.5e5f
#define HLEAK_C     0.0f
#define TAU_MEM_I   100.0f
#define TAU_SYN_I   200.0f

typedef __attribute__((ext_vector_type(16))) __bf16    v16bf;
typedef __attribute__((ext_vector_type(8)))  float     v8f;
typedef __attribute__((ext_vector_type(8)))  unsigned  v8u;

// Round-to-nearest-even f32 -> bf16, packed pair into one u32.
__device__ __forceinline__ unsigned bf16pk(float lo, float hi) {
  unsigned ul = __builtin_bit_cast(unsigned, lo);
  unsigned uh = __builtin_bit_cast(unsigned, hi);
  ul += 0x7fffu + ((ul >> 16) & 1u);
  uh += 0x7fffu + ((uh >> 16) & 1u);
  return (ul >> 16) | (uh & 0xffff0000u);
}

// B fragment: 8 consecutive u32 per lane (pre-swizzled) -> two b128 loads.
__device__ __forceinline__ v8u ld_frag(const unsigned* p) {
  v8u r;
  *reinterpret_cast<uint4*>(&r)       = *reinterpret_cast<const uint4*>(p);
  *(reinterpret_cast<uint4*>(&r) + 1) = *reinterpret_cast<const uint4*>(p + 4);
  return r;
}

// A fragment from a bf16 row (16 u32 per 32-k tile), contiguous k order to
// match the B swizzle: lanes 0-15 hold k 0..15; lanes 16-31 hold k 16..31.
// u32 cols: [8*hf .. 8*hf+7].
__device__ __forceinline__ v8u ld_afrag(const unsigned* rowp, int hf) {
  v8u r;
  *reinterpret_cast<uint4*>(&r)       = *reinterpret_cast<const uint4*>(rowp + 8 * hf);
  *(reinterpret_cast<uint4*>(&r) + 1) = *reinterpret_cast<const uint4*>(rowp + 4 + 8 * hf);
  return r;
}

__device__ __forceinline__ v8f wmma_bf16(v8u a, v8u b, v8f c) {
  return __builtin_amdgcn_wmma_f32_16x16x32_bf16(
      false, __builtin_bit_cast(v16bf, a),
      false, __builtin_bit_cast(v16bf, b),
      (short)0, c, false, false);
}

// Flat->LDS address: aperture mapping truncates to addr[31:0] (ISA 10.2).
__device__ __forceinline__ unsigned lds_addr(const void* p) {
  return (unsigned)(unsigned long long)p;
}

// Wave-private async DMA: wave wv streams its own 64-col f32 stripe of one
// [16 x 512] slice (16 rows x 256 B) into LDS. 256 chunks of 16 B per wave ->
// 8 b128 issues per lane. Tracked by ASYNCcnt; own-wave s_wait_asynccnt 0 is
// sufficient because only this wave ever reads this stripe.
__device__ __forceinline__ void async_ain_stripe(const float* gslice, float* lbuf,
                                                 int wv, int lane) {
  unsigned lbase = lds_addr(lbuf);
#pragma unroll
  for (int c = 0; c < 8; ++c) {
    int chunk = c * 32 + lane;                 // 0..255 within this wave
    int row = chunk >> 4, sub = chunk & 15;
    unsigned byteoff = (unsigned)(row * (HID * 4) + wv * 256 + sub * 16);
    unsigned laddr = lbase + byteoff;
    unsigned long long ga = (unsigned long long)gslice + byteoff;
    asm volatile("global_load_async_to_lds_b128 %0, %1, off"
                 :: "v"(laddr), "v"(ga) : "memory");
  }
}

__device__ __forceinline__ void wait_async0() {
  asm volatile("s_wait_asynccnt 0x0" ::: "memory");
}

// ---------------------------------------------------------------------------
// Phase 0: w (row-major [N][K], i.e. w[n*K+k]) -> bf16 B-fragment tiles.
// Tile (ktile, ntile) covers B[k0:k0+32, n0:n0+16] = w^T. Per-lane layout
// (B 32x16): lanes 0-15 -> n = lane, k rows 0..15 (VGPR v holds k=2v,2v+1);
// lanes 16-31 -> same n, k rows 16..31. Zero-padded at K and N tails.
// ---------------------------------------------------------------------------
__global__ __launch_bounds__(256) void swizzle_w(
    const float* __restrict__ w, unsigned* __restrict__ out,
    int N, int K, int ntiles, int nSlots)
{
  int idx = blockIdx.x * blockDim.x + threadIdx.x;
  if (idx >= nSlots) return;
  int v     = idx & 7;
  int lane  = (idx >> 3) & 31;
  int tile  = idx >> 8;
  int ktile = tile / ntiles, ntile = tile % ntiles;
  int hf    = lane >> 4;
  int n     = ntile * 16 + (lane & 15);
  int k     = ktile * 32 + 2 * v + 16 * hf;
  float lo = 0.f, hi = 0.f;
  if (n < N) {
    if (k     < K) lo = w[(size_t)n * K + k];
    if (k + 1 < K) hi = w[(size_t)n * K + k + 1];
  }
  out[idx] = bf16pk(lo, hi);
}

// ---------------------------------------------------------------------------
// Phase 1: C[M=T*BF, HID] = X[M, IN] * w_in^T, bf16 WMMA.
// Block tile 128(M) x 64(N); 8 waves: wave w owns rows [w*16, w*16+16) and
// all 4 n-subtiles. A staged in LDS as bf16 (row stride 24 u32 = 96 B,
// 16B-aligned, bank-spread).
// ---------------------------------------------------------------------------
__global__ __launch_bounds__(256) void in_gemm(
    const float* __restrict__ X, const unsigned* __restrict__ Bw,
    float* __restrict__ C, int K, int ktiles, int ntilesTot)
{
  __shared__ unsigned As[128][24];   // 32 bf16 used per row, padded stride

  const int tid  = threadIdx.x;
  const int lane = tid & 31, wv = tid >> 5, hf = lane >> 4;
  const int m16  = lane & 15;
  const int row0 = blockIdx.y * 128;
  const int nt0  = blockIdx.x * 4;   // first 16-wide n-subtile index

  const v8f vzero = {0.f, 0.f, 0.f, 0.f, 0.f, 0.f, 0.f, 0.f};
  v8f acc[4];
  for (int nt = 0; nt < 4; ++nt) acc[nt] = vzero;

  for (int kt = 0; kt < ktiles; ++kt) {
    const int k0 = kt * 32;
    // Cooperative, coalesced stage of A tile (f32 -> bf16 in LDS).
    for (int f4 = tid; f4 < 1024; f4 += 256) {
      int r = f4 >> 3, kq = f4 & 7;
      int kg = k0 + kq * 4;
      const float* xp = X + (size_t)(row0 + r) * K + kg;
      float x0 = 0.f, x1 = 0.f, x2 = 0.f, x3 = 0.f;
      if (kg + 3 < K) {
        float4 xv = *reinterpret_cast<const float4*>(xp);
        x0 = xv.x; x1 = xv.y; x2 = xv.z; x3 = xv.w;
      } else {
        if (kg     < K) x0 = xp[0];
        if (kg + 1 < K) x1 = xp[1];
        if (kg + 2 < K) x2 = xp[2];
        if (kg + 3 < K) x3 = xp[3];
      }
      As[r][kq * 2]     = bf16pk(x0, x1);
      As[r][kq * 2 + 1] = bf16pk(x2, x3);
    }
    // Prefetch next A tile (lowers to global_prefetch_b8).
    if (kt + 1 < ktiles)
      __builtin_prefetch(X + (size_t)(row0 + (tid >> 3)) * K + k0 + 32 + (tid & 7) * 4, 0, 0);
    __syncthreads();

    const v8u a = ld_afrag(As[wv * 16 + m16], hf);
#pragma unroll
    for (int nt = 0; nt < 4; ++nt) {
      const unsigned* bp = Bw + ((size_t)kt * ntilesTot + (nt0 + nt)) * 256 + lane * 8;
      acc[nt] = wmma_bf16(a, ld_frag(bp), acc[nt]);
    }
    __syncthreads();
  }

  // C layout: lane l, VGPR r -> (m = r + 8*(l>>4), n = l&15).
  for (int nt = 0; nt < 4; ++nt)
    for (int r = 0; r < 8; ++r) {
      int m = row0 + wv * 16 + r + 8 * hf;
      int n = (nt0 + nt) * 16 + m16;
      C[(size_t)m * HID + n] = acc[nt][r];
    }
}

// ---------------------------------------------------------------------------
// Phase 2: sequential scan. One block per 16 batch rows (self-contained:
// cur[b,h] depends only on z[b,:]).  z double-buffered in LDS (bf16, row
// stride 520 = 1040 B, 16B-aligned, bank-spread): step t reads zb[t&1],
// writes zb[(t+1)&1] -> one workgroup barrier per step. pos state lives in
// WMMA-C-layout VGPRs. Ain[t] stripes streamed by per-wave async DMA
// during step t's WMMAs. Wave 0 does the readout tile after the barrier,
// overlapped with the other waves' next-step GEMM.
// ---------------------------------------------------------------------------
__global__ __launch_bounds__(256) void dwmtj_scan(
    const float* __restrict__ Ain, const unsigned* __restrict__ Wrec,
    const unsigned* __restrict__ Wout, float* __restrict__ Vout)
{
  __shared__ unsigned zsh[2][16 * 260];       // 2 x (16 rows x 520 bf16)
  __shared__ float    ain_sh[2][16 * HID];    // double-buffered input slices

  const int tid  = threadIdx.x;
  const int lane = tid & 31, wv = tid >> 5, hf = lane >> 4;
  const int m16  = lane & 15;
  const int bt   = blockIdx.x;                // batch tile (16 rows)

  for (int i = tid; i < 16 * 260; i += 256) zsh[0][i] = 0u;   // z_{-1} = 0

  float pos[4][8];
  for (int j = 0; j < 4; ++j)
    for (int r = 0; r < 8; ++r) pos[j][r] = 0.f;
  float vmem[8], isyn[8];
  for (int r = 0; r < 8; ++r) { vmem[r] = 0.f; isyn[r] = 0.f; }

  const v8f vzero = {0.f, 0.f, 0.f, 0.f, 0.f, 0.f, 0.f, 0.f};

  // Kick off the t=0 input stripe DMA; completion checked inside the loop.
  async_ain_stripe(Ain + (size_t)(bt * 16) * HID, ain_sh[0], wv, lane);
  __syncthreads();

  for (int t = 0; t < T_STEPS; ++t) {
    const unsigned* zrd = zsh[t & 1];                          // z_{t-1}
    unsigned short* zwr = (unsigned short*)zsh[(t + 1) & 1];   // z_t

    // ---- recurrent GEMM against z_{t-1} (overlaps with Ain[t] DMA) -------
    v8f acc[4];
#pragma unroll
    for (int j = 0; j < 4; ++j) {
      const int ntile = wv * 4 + j;
      v8f a4 = vzero;
#pragma unroll
      for (int kt = 0; kt < 16; ++kt) {
        if (kt < 15)
          __builtin_prefetch(Wrec + ((size_t)(kt + 1) * 32 + ntile) * 256 + lane * 8, 0, 0);
        const v8u a = ld_afrag(zrd + m16 * 260 + kt * 16, hf);
        const v8u b = ld_frag(Wrec + ((size_t)kt * 32 + ntile) * 256 + lane * 8);
        a4 = wmma_bf16(a, b, a4);
      }
      acc[j] = a4;
    }

    wait_async0();                 // own wave's Ain[t] stripe is in LDS

    // Stream next step's stripe (reuses the buffer this wave consumed at t-1;
    // its reads were consumed before this point, same wave -> no hazard).
    if (t + 1 < T_STEPS)
      async_ain_stripe(Ain + ((size_t)(t + 1) * BF + bt * 16) * HID,
                       ain_sh[(t + 1) & 1], wv, lane);

    // ---- elementwise DWMTJ update; z_t into the other z buffer -----------
    const float* ain = ain_sh[t & 1];
#pragma unroll
    for (int j = 0; j < 4; ++j) {
      const int n = (wv * 4 + j) * 16 + m16;
#pragma unroll
      for (int r = 0; r < 8; ++r) {
        const int m = r + 8 * hf;
        float cur = acc[j][r] + ain[m * HID + n] + I_APP_C;
        float p   = pos[j][r] + DT_C * (MU_C * cur - HLEAK_C * pos[j][r]);
        float z   = (p - W2_C > 0.f) ? 1.f : 0.f;   // superspike fwd
        pos[j][r] = p * (1.f - z);                  // reset on fire
        zwr[m * 520 + n] = (z > 0.f) ? (unsigned short)0x3f80 : (unsigned short)0;
      }
    }

    __syncthreads();               // publish z_t; only barrier this step

    if (wv == 0) {                 // readout of z_t (N padded 10->16),
      v8f ro = vzero;              // overlaps other waves' next-step GEMM
      const unsigned* zro = zsh[(t + 1) & 1];
#pragma unroll
      for (int kt = 0; kt < 16; ++kt) {
        const v8u a = ld_afrag(zro + m16 * 260 + kt * 16, hf);
        const v8u b = ld_frag(Wout + (size_t)kt * 256 + lane * 8);
        ro = wmma_bf16(a, b, ro);
      }
#pragma unroll
      for (int r = 0; r < 8; ++r) {
        float vn = vmem[r] + DT_C * TAU_MEM_I * (-vmem[r] + isyn[r]);
        isyn[r]  = isyn[r] + DT_C * TAU_SYN_I * (-isyn[r]) + ro[r];
        vmem[r]  = vn;
        if (m16 < OUTF) {
          const int m = r + 8 * hf;
          Vout[((size_t)t * BF + bt * 16 + m) * OUTF + m16] = vn;
        }
      }
    }
  }
}

// ---------------------------------------------------------------------------
extern "C" void kernel_launch(void* const* d_in, const int* in_sizes, int n_in,
                              void* d_out, int out_size, void* d_ws, size_t ws_size,
                              hipStream_t stream) {
  (void)in_sizes; (void)n_in; (void)out_size; (void)ws_size;

  const float* x     = (const float*)d_in[0];   // [T,B,C,H,W] == [T*256, 2312]
  const float* w_in  = (const float*)d_in[1];   // [512, 2312]
  const float* w_rec = (const float*)d_in[2];   // [512, 512]
  const float* w_out = (const float*)d_in[3];   // [10, 512]
  float*       out   = (float*)d_out;           // [100, 256, 10]
  char*        ws    = (char*)d_ws;

  const int KT_IN  = (IN_F + 31) / 32;   // 73  (K padded to 2336)
  const int NT_H   = HID / 16;           // 32
  const int KT_REC = HID / 32;           // 16

  const int slots_win = KT_IN  * NT_H * 256;   // 598016 u32
  const int slots_rec = KT_REC * NT_H * 256;   // 131072 u32
  const int slots_out = KT_REC * 1    * 256;   // 4096   u32

  // Workspace layout (~55.4 MB total).
  size_t offA    = 0;
  size_t szA     = (size_t)T_STEPS * BF * HID * sizeof(float);   // 52.43 MB
  size_t offWin  = offA + szA;
  size_t szWin   = (size_t)slots_win * 4;
  size_t offWrec = offWin + szWin;
  size_t szWrec  = (size_t)slots_rec * 4;
  size_t offWout = offWrec + szWrec;

  float*    Ain = (float*)(ws + offA);
  unsigned* Win = (unsigned*)(ws + offWin);
  unsigned* Wrc = (unsigned*)(ws + offWrec);
  unsigned* Wot = (unsigned*)(ws + offWout);

  // Phase 0: weight swizzle/convert.
  swizzle_w<<<(slots_win + 255) / 256, 256, 0, stream>>>(w_in,  Win, HID,  IN_F, NT_H, slots_win);
  swizzle_w<<<(slots_rec + 255) / 256, 256, 0, stream>>>(w_rec, Wrc, HID,  HID,  NT_H, slots_rec);
  swizzle_w<<<(slots_out + 255) / 256, 256, 0, stream>>>(w_out, Wot, OUTF, HID,  1,    slots_out);

  // Phase 1: input projection for all timesteps (parallel across device).
  in_gemm<<<dim3(HID / 64, (T_STEPS * BF) / 128), 256, 0, stream>>>(x, Win, Ain, IN_F, KT_IN, NT_H);

  // Phase 2: sequential DWMTJ + LI scan.
  dwmtj_scan<<<BF / 16, 256, 0, stream>>>(Ain, Wrc, Wot, out);
}
